// EdgeGAT_19241453486701
// MI455X (gfx1250) — matrix operand order repaired and checked by reading, not compiled, not run on hardware
//
#include <hip/hip_runtime.h>
#include <hip/hip_bf16.h>

typedef __attribute__((ext_vector_type(16))) __bf16 v16bf;
typedef __attribute__((ext_vector_type(8)))  float  v8f;

#define GAT_H 8
#define LRELU 0.2f
#define NEGINF -1e9f

union BFrag {
    v16bf v;
    unsigned short u[16];
    unsigned int   w[8];
};

__device__ inline unsigned short f2bf(float f) {
    unsigned int u = __float_as_uint(f);
    unsigned int r = u + 0x7FFFu + ((u >> 16) & 1u);   // RNE
    return (unsigned short)(r >> 16);
}

__device__ inline void atomicMaxF(float* addr, float val) {
    unsigned int* ua = (unsigned int*)addr;
    unsigned int old = *ua;
    while (__uint_as_float(old) < val) {
        unsigned int assumed = old;
        old = atomicCAS(ua, assumed, __float_as_uint(val));
        if (old == assumed) break;
    }
}

// ---------------------------------------------------------------------------
// WMMA GEMM: Y[M, NT*16] = X[M,128] @ W[128, NT*16]   (bf16 inputs, f32 acc)
// One wave per 16-row stripe, 8 waves (256 thr) per block -> 128 rows/block.
// W is staged in LDS as bf16, transposed [col][k] so B pairs load as b32.
// ---------------------------------------------------------------------------
template <int NT>
__global__ void __launch_bounds__(256)
gat_gemm_wmma(const float* __restrict__ X, const float* __restrict__ W,
              float* __restrict__ Y, int M) {
    constexpr int Ncol = NT * 16;
    __shared__ unsigned short ldsW[128 * Ncol];

    for (int idx = threadIdx.x; idx < 128 * Ncol; idx += 256) {
        int k = idx / Ncol, col = idx % Ncol;
        ldsW[col * 128 + k] = f2bf(W[idx]);
    }
    __syncthreads();

    const int wave = threadIdx.x >> 5;
    const int lane = threadIdx.x & 31;
    const int half = lane >> 4;        // K half-select
    const int l16  = lane & 15;

    const long rowBase = (long)blockIdx.x * 128 + wave * 16;
    if (rowBase >= M) return;          // wave-uniform: EXEC stays all-ones

    v8f acc[NT] = {};
    const float* xrow = X + (rowBase + l16) * 128;

    #pragma unroll
    for (int kb = 0; kb < 4; ++kb) {   // 4 x K=32 steps = K=128
        BFrag a;
        #pragma unroll
        for (int v = 0; v < 4; ++v) {  // K = kb*32 + 8*half + {0..7}
            int k0 = kb * 32 + 8 * half + 2 * v;
            float2 f = *(const float2*)(xrow + k0);
            a.u[2 * v]     = f2bf(f.x);
            a.u[2 * v + 1] = f2bf(f.y);
        }
        #pragma unroll
        for (int v = 0; v < 4; ++v) {  // K = kb*32 + 16 + 8*half + {0..7}
            int k0 = kb * 32 + 16 + 8 * half + 2 * v;
            float2 f = *(const float2*)(xrow + k0);
            a.u[8 + 2 * v]     = f2bf(f.x);
            a.u[8 + 2 * v + 1] = f2bf(f.y);
        }
        #pragma unroll
        for (int nt = 0; nt < NT; ++nt) {
            BFrag b;
            int col = nt * 16 + l16;
            #pragma unroll
            for (int v = 0; v < 8; ++v) {   // K = kb*32 + 16*(v/4) + 8*half + 2*(v%4) + {0,1}
                int k = kb * 32 + 16 * (v >> 2) + 8 * half + 2 * (v & 3);
                b.w[v] = *(const unsigned int*)&ldsW[col * 128 + k];
            }
            acc[nt] = __builtin_amdgcn_wmma_f32_16x16x32_bf16(
                false, a.v, false, b.v, (short)0, acc[nt], false, false);
        }
    }

    #pragma unroll
    for (int nt = 0; nt < NT; ++nt) {
        #pragma unroll
        for (int r = 0; r < 8; ++r) {
            long row = rowBase + r + 8 * half;
            Y[row * Ncol + nt * 16 + l16] = acc[nt][r];
        }
    }
}

// s_src/s_dst[n,h] = sum_c h[n,h,c] * a_{src,dst}[h,c]
__global__ void gat_scores(const float* __restrict__ h,
                           const float* __restrict__ a_src,
                           const float* __restrict__ a_dst,
                           float* __restrict__ ssrc, float* __restrict__ sdst,
                           int N_, int C) {
    int idx = blockIdx.x * blockDim.x + threadIdx.x;  // over N*H
    if (idx >= N_ * GAT_H) return;
    int n = idx / GAT_H, hh = idx % GAT_H;
    const float* hv = h + (long)n * (GAT_H * C) + hh * C;
    const float* as = a_src + hh * C;
    const float* ad = a_dst + hh * C;
    float s1 = 0.f, s2 = 0.f;
    for (int c = 0; c < C; ++c) { float x = hv[c]; s1 += x * as[c]; s2 += x * ad[c]; }
    ssrc[idx] = s1; sdst[idx] = s2;
}

__global__ void gat_init(float* __restrict__ m, float* __restrict__ denom,
                         float* __restrict__ agg, int NH, long NHC) {
    long i = blockIdx.x * (long)blockDim.x + threadIdx.x;
    if (i < NH) { m[i] = NEGINF; denom[i] = 0.f; }
    if (i < NHC) agg[i] = 0.f;
}

// segment_max of leaky-relu logits over destinations (edges of type t + loops)
__global__ void gat_edge_max(const float* __restrict__ ssrc, const float* __restrict__ sdst,
                             const int* __restrict__ esrc, const int* __restrict__ edst,
                             const int* __restrict__ etype, int t, int E_, int N_,
                             float* __restrict__ m) {
    int i = blockIdx.x * blockDim.x + threadIdx.x;
    if (i >= E_ + N_) return;
    int s, d;
    if (i < E_) { if (etype[i] != t) return; s = esrc[i]; d = edst[i]; }
    else        { s = d = i - E_; }
    const float* ps = ssrc + (long)s * GAT_H;
    const float* pd = sdst + (long)d * GAT_H;
    float* pm = m + (long)d * GAT_H;
    #pragma unroll
    for (int hh = 0; hh < GAT_H; ++hh) {
        float e = ps[hh] + pd[hh];
        e = e > 0.f ? e : LRELU * e;
        atomicMaxF(&pm[hh], e);
    }
}

// p = exp(e - m[dst]); denom += p; agg[dst] += p * h[src]
__global__ void gat_edge_agg(const float* __restrict__ ssrc, const float* __restrict__ sdst,
                             const float* __restrict__ m,
                             const int* __restrict__ esrc, const int* __restrict__ edst,
                             const int* __restrict__ etype, int t, int E_, int N_,
                             const float* __restrict__ h,
                             float* __restrict__ denom, float* __restrict__ agg, int C) {
    long tid = blockIdx.x * (long)blockDim.x + threadIdx.x;
    if (tid >= (long)(E_ + N_) * GAT_H) return;
    int i  = (int)(tid >> 3);
    int hh = (int)(tid & 7);
    int s, d;
    if (i < E_) { if (etype[i] != t) return; s = esrc[i]; d = edst[i]; }
    else        { s = d = i - E_; }
    float e = ssrc[(long)s * GAT_H + hh] + sdst[(long)d * GAT_H + hh];
    e = e > 0.f ? e : LRELU * e;
    float p = __expf(e - m[(long)d * GAT_H + hh]);
    atomicAdd(&denom[(long)d * GAT_H + hh], p);
    const int HC = GAT_H * C;
    const float* hv = h + (long)s * HC + hh * C;
    float* av = agg + (long)d * HC + hh * C;
    for (int c = 0; c < C; ++c) atomicAdd(&av[c], p * hv[c]);
}

// out = agg/denom + bias (optional relu), strided/offset write
__global__ void gat_finalize(const float* __restrict__ agg, const float* __restrict__ denom,
                             const float* __restrict__ b, float* __restrict__ out,
                             int N_, int C, int ldOut, int colOff, int doRelu) {
    const int HC = GAT_H * C;
    long j = blockIdx.x * (long)blockDim.x + threadIdx.x;
    if (j >= (long)N_ * HC) return;
    int n   = (int)(j / HC);
    int rem = (int)(j % HC);
    int hh  = rem / C;
    float v = agg[j] / denom[(long)n * GAT_H + hh] + b[rem];
    if (doRelu) v = v > 0.f ? v : 0.f;
    out[(long)n * ldOut + colOff + rem] = v;
}

extern "C" void kernel_launch(void* const* d_in, const int* in_sizes, int n_in,
                              void* d_out, int out_size, void* d_ws, size_t ws_size,
                              hipStream_t stream) {
    const float* x     = (const float*)d_in[0];
    const int*   esrc  = (const int*)d_in[1];
    const int*   edst  = (const int*)d_in[2];
    const int*   etype = (const int*)d_in[3];
    const float* W1    = (const float*)d_in[4];   // [T,128,128]
    const float* a1s   = (const float*)d_in[5];   // [T,8,16]
    const float* a1d   = (const float*)d_in[6];
    const float* b1    = (const float*)d_in[7];   // [T,128]
    const float* W2    = (const float*)d_in[8];   // [T,128,64]
    const float* a2s   = (const float*)d_in[9];   // [T,8,8]
    const float* a2d   = (const float*)d_in[10];
    const float* b2    = (const float*)d_in[11];  // [T,64]
    float* out = (float*)d_out;

    const int N = in_sizes[0] / 128;   // 50000
    const int E = in_sizes[1];         // 800000
    const int T = 4;

    float* ws    = (float*)d_ws;
    float* proj  = ws;                       // [N,128] current projection
    float* conv1 = ws + (long)N * 128;       // [N,128] layer-1 output (relu)
    float* agg   = ws + (long)N * 256;       // [N,128] aggregation scratch
    float* ssrc  = ws + (long)N * 384;       // [N,8]
    float* sdst  = ssrc + (long)N * 8;
    float* mbuf  = sdst + (long)N * 8;
    float* denom = mbuf + (long)N * 8;

    const int EN = E + N;
    const dim3 blk(256);
    const int gGemm  = (N + 127) / 128;
    const int gNH    = (N * GAT_H + 255) / 256;
    const int gE     = (EN + 255) / 256;
    const int gEH    = (int)(((long)EN * GAT_H + 255) / 256);
    const int gNHC1  = (N * 128 + 255) / 256;
    const int gNHC2  = (N * 64 + 255) / 256;

    for (int t = 0; t < T; ++t) {
        // ---- layer 1 (C=16, Ncol=128) ----
        gat_gemm_wmma<8><<<gGemm, blk, 0, stream>>>(x, W1 + (long)t * 128 * 128, proj, N);
        gat_scores<<<gNH, blk, 0, stream>>>(proj, a1s + t * 128, a1d + t * 128,
                                            ssrc, sdst, N, 16);
        gat_init<<<gNHC1, blk, 0, stream>>>(mbuf, denom, agg, N * GAT_H, (long)N * 128);
        gat_edge_max<<<gE, blk, 0, stream>>>(ssrc, sdst, esrc, edst, etype, t, E, N, mbuf);
        gat_edge_agg<<<gEH, blk, 0, stream>>>(ssrc, sdst, mbuf, esrc, edst, etype, t, E, N,
                                              proj, denom, agg, 16);
        gat_finalize<<<gNHC1, blk, 0, stream>>>(agg, denom, b1 + t * 128, conv1,
                                                N, 16, 128, 0, 1);
        // ---- layer 2 (C=8, Ncol=64) ----
        gat_gemm_wmma<4><<<gGemm, blk, 0, stream>>>(conv1, W2 + (long)t * 128 * 64, proj, N);
        gat_scores<<<gNH, blk, 0, stream>>>(proj, a2s + t * 64, a2d + t * 64,
                                            ssrc, sdst, N, 8);
        gat_init<<<gNHC2, blk, 0, stream>>>(mbuf, denom, agg, N * GAT_H, (long)N * 64);
        gat_edge_max<<<gE, blk, 0, stream>>>(ssrc, sdst, esrc, edst, etype, t, E, N, mbuf);
        gat_edge_agg<<<gEH, blk, 0, stream>>>(ssrc, sdst, mbuf, esrc, edst, etype, t, E, N,
                                              proj, denom, agg, 8);
        gat_finalize<<<gNHC2, blk, 0, stream>>>(agg, denom, b2 + t * 64, out,
                                                N, 8, 256, t * 64, 0);
    }
}
